// BionaEmformer_35605278884597
// MI455X (gfx1250) — matrix operand order repaired and checked by reading, not compiled, not run on hardware
//
#include <hip/hip_runtime.h>
#include <hip/hip_bf16.h>

// ---------------- Fixed Emformer config ----------------
#define B_    256
#define T_    2
#define D_    80
#define AD_   96            // padded row stride for D-wide activation buffers (mult of 32)
#define H_    8
#define HD_   10
#define LC_   64
#define M_    4
#define L_    20
#define FFN_  2048
#define V_    29
#define EMB_  512
#define KT_   (M_ + LC_ + T_)        // 70 keys
#define SK_   8                      // split-K factor for the K=2048 GEMM
#define SCALE_ 0.31622776601683794f  // 10^-0.5
#define NEGINF_ (-1e8f)

typedef __attribute__((ext_vector_type(16))) _Float16 v16h;
typedef __attribute__((ext_vector_type(8)))  float    v8f;
typedef __attribute__((ext_vector_type(4)))  float    f32x4;
typedef __attribute__((ext_vector_type(8)))  _Float16 h16x8;

static inline int cdiv(int a, int b) { return (a + b - 1) / b; }
static inline int rup32(int a) { return (a + 31) & ~31; }

// ---------------- zero fill ----------------
__global__ void zero_kernel(float* __restrict__ p, int n) {
    int i = blockIdx.x * blockDim.x + threadIdx.x;
    if (i < n) p[i] = 0.0f;
}

// ---------------- f32 -> f16 weight conversion with N/K zero padding ----------------
// src: [Lc, N, K] f32   dst: [Lc, Np, Kp] f16 (zero padded)
__global__ void cvt_pad_kernel(const float* __restrict__ s, _Float16* __restrict__ d,
                               int Lc, int N, int K, int Np, int Kp) {
    int i = blockIdx.x * blockDim.x + threadIdx.x;
    int tot = Lc * Np * Kp;
    if (i >= tot) return;
    int k = i % Kp;
    int n = (i / Kp) % Np;
    int l = i / (Kp * Np);
    _Float16 v = (_Float16)0.0f;
    if (n < N && k < K) v = (_Float16)s[((size_t)l * N + n) * K + k];
    d[i] = v;
}

// ---------------- shared WMMA tile body ----------------
// One wave computes a 16x32 C macro-tile over K range [kb, ke): two v_wmma per K-step.
// All strides/pads arranged so the loop has zero predication; fragments via b128 loads.
__device__ __forceinline__ void wmma_tile(const float* __restrict__ A, int lda,
                                          const _Float16* __restrict__ W, int ldb,
                                          int tm, int tn, int kb, int ke, int lane,
                                          v8f& acc0, v8f& acc1) {
    const int l15 = lane & 15;
    const int kbA = (lane >> 4) * 8;         // A lane K-base {0,8}
    const int kbB = (lane >> 4) * 8;         // B lane K-base {0,8}

    const float*    Ap  = A + (size_t)(tm + l15) * lda;
    const _Float16* Bp0 = W + (size_t)(tn + l15) * ldb;
    const _Float16* Bp1 = Bp0 + (size_t)16 * ldb;

    for (int k0 = kb; k0 < ke; k0 += 32) {
        const float* pa0 = Ap + k0 + kbA;
        const float* pa1 = Ap + k0 + 16 + kbA;
        f32x4 a00 = *(const f32x4*)(pa0);
        f32x4 a01 = *(const f32x4*)(pa0 + 4);
        f32x4 a10 = *(const f32x4*)(pa1);
        f32x4 a11 = *(const f32x4*)(pa1 + 4);
        v16h a;
#pragma unroll
        for (int j = 0; j < 4; ++j) {
            a[j]      = (_Float16)a00[j];
            a[j + 4]  = (_Float16)a01[j];
            a[j + 8]  = (_Float16)a10[j];
            a[j + 12] = (_Float16)a11[j];
        }
        h16x8 b0lo = *(const h16x8*)(Bp0 + k0 + kbB);
        h16x8 b0hi = *(const h16x8*)(Bp0 + k0 + 16 + kbB);
        h16x8 b1lo = *(const h16x8*)(Bp1 + k0 + kbB);
        h16x8 b1hi = *(const h16x8*)(Bp1 + k0 + 16 + kbB);
        v16h b0, b1;
#pragma unroll
        for (int j = 0; j < 8; ++j) {
            b0[j] = b0lo[j]; b0[j + 8] = b0hi[j];
            b1[j] = b1lo[j]; b1[j + 8] = b1hi[j];
        }
        acc0 = __builtin_amdgcn_wmma_f32_16x16x32_f16(false, a, false, b0,
                                                      (short)0, acc0, false, false);
        acc1 = __builtin_amdgcn_wmma_f32_16x16x32_f16(false, a, false, b1,
                                                      (short)0, acc1, false, false);
    }
}

// ---------------- full-K WMMA GEMM: C[M,N] = A[M,K](f32,lda) * W[N,K](f16,ldb) + bias ----------------
__global__ void wmma_gemm_kernel(const float* __restrict__ A, int lda,
                                 const _Float16* __restrict__ W, int ldb,
                                 const float* __restrict__ bias,
                                 float* __restrict__ C, int ldc,
                                 int Nc, int Kp, int relu) {
    const int lane = threadIdx.x;
    const int tm = blockIdx.y * 16;
    const int tn = blockIdx.x * 32;
    v8f acc0 = {}, acc1 = {};
    wmma_tile(A, lda, W, ldb, tm, tn, 0, Kp, lane, acc0, acc1);

    // C layout: lanes 0-15 -> N=lane, M=vgpr; lanes 16-31 -> N=lane-16, M=vgpr+8
    const int rbase = (lane >> 4) * 8;
    const int n0 = tn + (lane & 15);
    const int n1 = n0 + 16;
    if (n0 < Nc) {
        float bv = bias ? bias[n0] : 0.0f;
#pragma unroll
        for (int r = 0; r < 8; ++r) {
            float v = acc0[r] + bv;
            if (relu) v = fmaxf(v, 0.0f);
            C[(size_t)(tm + rbase + r) * ldc + n0] = v;
        }
    }
    if (n1 < Nc) {
        float bv = bias ? bias[n1] : 0.0f;
#pragma unroll
        for (int r = 0; r < 8; ++r) {
            float v = acc1[r] + bv;
            if (relu) v = fmaxf(v, 0.0f);
            C[(size_t)(tm + rbase + r) * ldc + n1] = v;
        }
    }
}

// ---------------- split-K WMMA GEMM: partial P[z][M,ldc] over K chunk z ----------------
__global__ void wmma_gemm_splitk_kernel(const float* __restrict__ A, int lda,
                                        const _Float16* __restrict__ W, int ldb,
                                        float* __restrict__ P, int ldc,
                                        int Mr, int Nc, int kChunk) {
    const int lane = threadIdx.x;
    const int tm = blockIdx.y * 16;
    const int tn = blockIdx.x * 32;
    const int z = blockIdx.z;
    v8f acc0 = {}, acc1 = {};
    wmma_tile(A, lda, W, ldb, tm, tn, z * kChunk, (z + 1) * kChunk, lane, acc0, acc1);

    float* C = P + (size_t)z * Mr * ldc;
    const int rbase = (lane >> 4) * 8;
    const int n0 = tn + (lane & 15);
    const int n1 = n0 + 16;
    if (n0 < Nc) {
#pragma unroll
        for (int r = 0; r < 8; ++r)
            C[(size_t)(tm + rbase + r) * ldc + n0] = acc0[r];
    }
    if (n1 < Nc) {
#pragma unroll
        for (int r = 0; r < 8; ++r)
            C[(size_t)(tm + rbase + r) * ldc + n1] = acc1[r];
    }
}

// deterministic split-K reduction: C[row,d] = bias[d] + sum_z P[z][row,d]
__global__ void splitk_reduce_kernel(const float* __restrict__ P, const float* __restrict__ bias,
                                     float* __restrict__ C, int rows) {
    int i = blockIdx.x * blockDim.x + threadIdx.x;
    if (i >= rows * D_) return;
    int d = i % D_; int row = i / D_;
    float s = bias[d];
#pragma unroll
    for (int z = 0; z < SK_; ++z) s += P[((size_t)z * rows + row) * AD_ + d];
    C[row * AD_ + d] = s;
}

// ---------------- LayerNorm over D=80, one wave per row (stride AD_) ----------------
__global__ void ln_kernel(const float* __restrict__ src, const float* __restrict__ src2,
                          const float* __restrict__ g, const float* __restrict__ bt,
                          float* __restrict__ dst0, float* __restrict__ dst1) {
    const int row = blockIdx.x;
    const int t = threadIdx.x;  // 32
    __shared__ float red[32];
    __shared__ float vals[D_];

    float local = 0.0f;
    for (int d = t; d < D_; d += 32) {
        float v = src[row * AD_ + d];
        if (src2) v += src2[row * AD_ + d];
        vals[d] = v;
        local += v;
    }
    red[t] = local; __syncthreads();
    for (int off = 16; off > 0; off >>= 1) { if (t < off) red[t] += red[t + off]; __syncthreads(); }
    float mu = red[0] / (float)D_;
    __syncthreads();

    float lv = 0.0f;
    for (int d = t; d < D_; d += 32) { float c = vals[d] - mu; lv += c * c; }
    red[t] = lv; __syncthreads();
    for (int off = 16; off > 0; off >>= 1) { if (t < off) red[t] += red[t + off]; __syncthreads(); }
    float inv = rsqrtf(red[0] / (float)D_ + 1e-5f);
    __syncthreads();

    for (int d = t; d < D_; d += 32) {
        float o = (vals[d] - mu) * inv * g[d] + bt[d];
        dst0[row * AD_ + d] = o;
        if (dst1) dst1[row * AD_ + d] = o;
    }
}

// ---------------- small elementwise kernels (stride AD_ for D-wide buffers) ----------------
__global__ void transpose_in_kernel(const float* __restrict__ chunk, float* __restrict__ x) {
    int i = blockIdx.x * blockDim.x + threadIdx.x;
    if (i >= T_ * B_ * D_) return;
    int d = i % D_; int b = (i / D_) % B_; int t = i / (D_ * B_);
    x[(t * B_ + b) * AD_ + d] = chunk[(b * T_ + t) * D_ + d];
}

__global__ void kvin_init_kernel(const float* __restrict__ memory, float* __restrict__ kvin) {
    int i = blockIdx.x * blockDim.x + threadIdx.x;
    if (i >= M_ * B_ * D_) return;
    int d = i % D_; int b = (i / D_) % B_; int m = i / (D_ * B_);
    kvin[(m * B_ + b) * AD_ + d] = memory[(b * M_ + m) * D_ + d];
}

__global__ void summary_kernel(float* __restrict__ qin) {
    int i = blockIdx.x * blockDim.x + threadIdx.x;
    if (i >= B_ * D_) return;
    int d = i % D_; int b = i / D_;
    qin[(2 * B_ + b) * AD_ + d] = 0.5f * (qin[b * AD_ + d] + qin[(B_ + b) * AD_ + d]);
}

__global__ void resid_kernel(const float* __restrict__ rc, const float* __restrict__ x,
                             float* __restrict__ res) {
    int i = blockIdx.x * blockDim.x + threadIdx.x;
    if (i >= T_ * B_ * D_) return;
    int d = i % D_; int row = i / D_;
    res[row * AD_ + d] = rc[row * AD_ + d] + x[row * AD_ + d];
}

__global__ void memclip_kernel(const float* __restrict__ rc, float* __restrict__ memclip) {
    int i = blockIdx.x * blockDim.x + threadIdx.x;
    if (i >= B_ * D_) return;
    int d = i % D_; int b = i / D_;
    float v = rc[(T_ * B_ + b) * AD_ + d];
    memclip[b * AD_ + d] = fminf(fmaxf(v, -10.0f), 10.0f);
}

__global__ void encmean_kernel(const float* __restrict__ x, float* __restrict__ em) {
    int i = blockIdx.x * blockDim.x + threadIdx.x;
    if (i >= B_ * D_) return;
    int d = i % D_; int b = i / D_;
    em[b * AD_ + d] = 0.5f * (x[b * AD_ + d] + x[(B_ + b) * AD_ + d]);
}

__global__ void memory_out_kernel(const float* __restrict__ memory,
                                  const float* __restrict__ memclip,
                                  float* __restrict__ out) {
    int i = blockIdx.x * blockDim.x + threadIdx.x;
    if (i >= B_ * M_ * D_) return;
    int d = i % D_; int m = (i / D_) % M_; int b = i / (D_ * M_);
    out[i] = (m < M_ - 1) ? memory[(b * M_ + m + 1) * D_ + d] : memclip[b * AD_ + d];
}

// rolling LC cache outputs (kvbuf stride 2*D_)
__global__ void lcout_kernel(const float* __restrict__ lck, const float* __restrict__ lcv,
                             const float* __restrict__ kvbuf,
                             float* __restrict__ outK, float* __restrict__ outV) {
    int i = blockIdx.x * blockDim.x + threadIdx.x;
    if (i >= B_ * LC_ * D_) return;
    int d = i % D_; int c = (i / D_) % LC_; int b = i / (D_ * LC_);
    float kv, vv;
    if (c < LC_ - T_) {
        kv = lck[(b * LC_ + c + T_) * D_ + d];
        vv = lcv[(b * LC_ + c + T_) * D_ + d];
    } else {
        int t = c - (LC_ - T_);
        kv = kvbuf[(size_t)((M_ + t) * B_ + b) * (2 * D_) + d];
        vv = kvbuf[(size_t)((M_ + t) * B_ + b) * (2 * D_) + D_ + d];
    }
    outK[i] = kv;
    outV[i] = vv;
}

// ---------------- attention: one thread per (q, b, h) ----------------
__global__ void attn_kernel(const float* __restrict__ qbuf,   // [3*B, AD_]
                            const float* __restrict__ kvbuf,  // [6*B, 2D]  (k | v)
                            const float* __restrict__ lck,    // [B, LC, D]
                            const float* __restrict__ lcv,
                            float* __restrict__ att) {        // [3*B, AD_]
    int idx = blockIdx.x * blockDim.x + threadIdx.x;
    if (idx >= (T_ + 1) * B_ * H_) return;
    int h = idx % H_;
    int b = (idx / H_) % B_;
    int q = idx / (H_ * B_);

    const float* qv = qbuf + (size_t)(q * B_ + b) * AD_ + h * HD_;
    float qr[HD_];
#pragma unroll
    for (int e = 0; e < HD_; ++e) qr[e] = qv[e] * SCALE_;

    float s[KT_];
    float mx = -3.0e38f;
    for (int kk = 0; kk < KT_; ++kk) {
        const float* kp;
        if (kk < M_)            kp = kvbuf + (size_t)(kk * B_ + b) * (2 * D_) + h * HD_;
        else if (kk < M_ + LC_) kp = lck + (size_t)(b * LC_ + (kk - M_)) * D_ + h * HD_;
        else                    kp = kvbuf + (size_t)((M_ + kk - M_ - LC_) * B_ + b) * (2 * D_) + h * HD_;
        float dsum = 0.0f;
#pragma unroll
        for (int e = 0; e < HD_; ++e) dsum += qr[e] * kp[e];
        if (q == T_ && kk < M_) dsum = NEGINF_;   // summary query masks the memory bank
        s[kk] = dsum;
        mx = fmaxf(mx, dsum);
    }
    float sum = 0.0f;
    for (int kk = 0; kk < KT_; ++kk) { s[kk] = __expf(s[kk] - mx); sum += s[kk]; }
    float inv = 1.0f / sum;

    float o[HD_];
#pragma unroll
    for (int e = 0; e < HD_; ++e) o[e] = 0.0f;
    for (int kk = 0; kk < KT_; ++kk) {
        const float* vp;
        if (kk < M_)            vp = kvbuf + (size_t)(kk * B_ + b) * (2 * D_) + D_ + h * HD_;
        else if (kk < M_ + LC_) vp = lcv + (size_t)(b * LC_ + (kk - M_)) * D_ + h * HD_;
        else                    vp = kvbuf + (size_t)((M_ + kk - M_ - LC_) * B_ + b) * (2 * D_) + D_ + h * HD_;
        float p = s[kk] * inv;
#pragma unroll
        for (int e = 0; e < HD_; ++e) o[e] += p * vp[e];
    }
    float* op = att + (size_t)(q * B_ + b) * AD_ + h * HD_;
#pragma unroll
    for (int e = 0; e < HD_; ++e) op[e] = o[e];
}

// ---------------- launch ----------------
extern "C" void kernel_launch(void* const* d_in, const int* in_sizes, int n_in,
                              void* d_out, int out_size, void* d_ws, size_t ws_size,
                              hipStream_t stream) {
    const float* chunk   = (const float*)d_in[0];
    const float* memory  = (const float*)d_in[1];
    const float* lc_key  = (const float*)d_in[2];
    const float* lc_val  = (const float*)d_in[3];
    const float* ln_in_g = (const float*)d_in[4];
    const float* ln_in_b = (const float*)d_in[5];
    const float* Wq      = (const float*)d_in[6];
    const float* bq      = (const float*)d_in[7];
    const float* Wkv     = (const float*)d_in[8];
    const float* bkv     = (const float*)d_in[9];
    const float* Wo      = (const float*)d_in[10];
    const float* bo      = (const float*)d_in[11];
    const float* ln_ff_g = (const float*)d_in[12];
    const float* ln_ff_b = (const float*)d_in[13];
    const float* W1      = (const float*)d_in[14];
    const float* b1      = (const float*)d_in[15];
    const float* W2      = (const float*)d_in[16];
    const float* b2      = (const float*)d_in[17];
    const float* ln_o_g  = (const float*)d_in[18];
    const float* ln_o_b  = (const float*)d_in[19];
    const float* ctc_W   = (const float*)d_in[20];
    const float* ctc_b   = (const float*)d_in[21];
    const float* emb_W   = (const float*)d_in[22];
    const float* emb_b   = (const float*)d_in[23];

    // -------- output regions (f32, concatenated in return order) --------
    float* out_logits = (float*)d_out;                                   // [T,B,V]
    float* out_mem    = out_logits + (size_t)T_ * B_ * V_;               // [B,M,D]
    float* out_lck    = out_mem + (size_t)B_ * M_ * D_;                  // [L,B,LC,D]
    float* out_lcv    = out_lck + (size_t)L_ * B_ * LC_ * D_;            // [L,B,LC,D]
    float* out_emb    = out_lcv + (size_t)L_ * B_ * LC_ * D_;            // [B,EMB]

    // -------- workspace carve --------
    char* ws = (char*)d_ws;
    size_t off = 0;
    auto carve = [&](size_t bytes) -> char* {
        char* p = ws + off;
        off = (off + bytes + 255) & ~(size_t)255;
        return p;
    };

    // padded f16 weights: [L, Np, Kp]
    const int DP = rup32(D_);        // 96
    const int VP = rup32(V_);        // 32
    _Float16* wq16  = (_Float16*)carve(sizeof(_Float16) * (size_t)L_ * DP * DP);
    _Float16* wkv16 = (_Float16*)carve(sizeof(_Float16) * (size_t)L_ * (2 * D_) * DP);
    _Float16* wo16  = (_Float16*)carve(sizeof(_Float16) * (size_t)L_ * DP * DP);
    _Float16* w116  = (_Float16*)carve(sizeof(_Float16) * (size_t)L_ * FFN_ * DP);
    _Float16* w216  = (_Float16*)carve(sizeof(_Float16) * (size_t)L_ * DP * FFN_);
    _Float16* ctc16 = (_Float16*)carve(sizeof(_Float16) * (size_t)VP * DP);
    _Float16* emb16 = (_Float16*)carve(sizeof(_Float16) * (size_t)EMB_ * DP);

    // f32 activations (stride AD_ for D-wide buffers) -- zero-filled region
    size_t act0 = off;
    float* x      = (float*)carve(sizeof(float) * (size_t)T_ * B_ * AD_);
    float* qin    = (float*)carve(sizeof(float) * (size_t)(T_ + 1) * B_ * AD_);
    float* kvin   = (float*)carve(sizeof(float) * (size_t)(M_ + T_) * B_ * AD_);
    float* qbuf   = (float*)carve(sizeof(float) * (size_t)(T_ + 1) * B_ * AD_);
    float* kvbuf  = (float*)carve(sizeof(float) * (size_t)(M_ + T_) * B_ * 2 * D_);
    float* attb   = (float*)carve(sizeof(float) * (size_t)(T_ + 1) * B_ * AD_);
    float* rc     = (float*)carve(sizeof(float) * (size_t)(T_ + 1) * B_ * AD_);
    float* res    = (float*)carve(sizeof(float) * (size_t)T_ * B_ * AD_);
    float* ffa    = (float*)carve(sizeof(float) * (size_t)T_ * B_ * AD_);
    float* hbuf   = (float*)carve(sizeof(float) * (size_t)T_ * B_ * FFN_);
    float* ffout  = (float*)carve(sizeof(float) * (size_t)T_ * B_ * AD_);
    float* pbuf   = (float*)carve(sizeof(float) * (size_t)SK_ * T_ * B_ * AD_);  // split-K partials
    float* memcl  = (float*)carve(sizeof(float) * (size_t)B_ * AD_);
    float* encm   = (float*)carve(sizeof(float) * (size_t)B_ * AD_);
    size_t actBytes = off - act0;

    // -------- zero activation region (pads must be zero; ws is poisoned) --------
    int actN = (int)(actBytes / sizeof(float));
    zero_kernel<<<cdiv(actN, 256), 256, 0, stream>>>((float*)(ws + act0), actN);

    // -------- weight conversion with padding (per launch; deterministic) --------
    auto cvt = [&](const float* src, _Float16* dst, int Lc, int N, int K, int Np, int Kp) {
        int tot = Lc * Np * Kp;
        cvt_pad_kernel<<<cdiv(tot, 256), 256, 0, stream>>>(src, dst, Lc, N, K, Np, Kp);
    };
    cvt(Wq,    wq16,  L_, D_,     D_,   DP,     DP);
    cvt(Wkv,   wkv16, L_, 2 * D_, D_,   2 * D_, DP);
    cvt(Wo,    wo16,  L_, D_,     D_,   DP,     DP);
    cvt(W1,    w116,  L_, FFN_,   D_,   FFN_,   DP);
    cvt(W2,    w216,  L_, D_,     FFN_, DP,     FFN_);
    cvt(ctc_W, ctc16, 1,  V_,     D_,   VP,     DP);
    cvt(emb_W, emb16, 1,  EMB_,   D_,   EMB_,   DP);

    // -------- init state --------
    transpose_in_kernel<<<cdiv(T_ * B_ * D_, 256), 256, 0, stream>>>(chunk, x);
    kvin_init_kernel<<<cdiv(M_ * B_ * D_, 256), 256, 0, stream>>>(memory, kvin);

    auto gemm = [&](const float* A, int lda, const _Float16* Wt, int ldb, const float* bias,
                    float* C, int ldc, int Mr, int Nc, int Kp, int relu) {
        dim3 grid(cdiv(Nc, 32), Mr / 16);
        wmma_gemm_kernel<<<grid, 32, 0, stream>>>(A, lda, Wt, ldb, bias, C, ldc, Nc, Kp, relu);
    };

    // -------- layer loop --------
    for (int l = 0; l < L_; ++l) {
        const float* lck_l = lc_key + (size_t)l * B_ * LC_ * D_;
        const float* lcv_l = lc_val + (size_t)l * B_ * LC_ * D_;

        // ln_in: ln_x into qin rows [0,2B) and kvin rows [4B,6B)
        ln_kernel<<<T_ * B_, 32, 0, stream>>>(x, nullptr,
                                              ln_in_g + l * D_, ln_in_b + l * D_,
                                              qin, kvin + (size_t)M_ * B_ * AD_);
        summary_kernel<<<cdiv(B_ * D_, 256), 256, 0, stream>>>(qin);

        gemm(qin,  AD_, wq16  + (size_t)l * DP * DP,     DP, bq  + l * D_,
             qbuf,  AD_,    (T_ + 1) * B_, D_,     DP, 0);
        gemm(kvin, AD_, wkv16 + (size_t)l * 2 * D_ * DP, DP, bkv + l * 2 * D_,
             kvbuf, 2 * D_, (M_ + T_) * B_, 2 * D_, DP, 0);

        attn_kernel<<<cdiv((T_ + 1) * B_ * H_, 64), 64, 0, stream>>>(qbuf, kvbuf, lck_l, lcv_l, attb);

        lcout_kernel<<<cdiv(B_ * LC_ * D_, 256), 256, 0, stream>>>(
            lck_l, lcv_l, kvbuf,
            out_lck + (size_t)l * B_ * LC_ * D_,
            out_lcv + (size_t)l * B_ * LC_ * D_);

        gemm(attb, AD_, wo16 + (size_t)l * DP * DP, DP, bo + l * D_,
             rc, AD_, (T_ + 1) * B_, D_, DP, 0);

        resid_kernel<<<cdiv(T_ * B_ * D_, 256), 256, 0, stream>>>(rc, x, res);
        if (l == L_ - 2) {
            memclip_kernel<<<cdiv(B_ * D_, 256), 256, 0, stream>>>(rc, memcl);
        }

        ln_kernel<<<T_ * B_, 32, 0, stream>>>(res, nullptr,
                                              ln_ff_g + l * D_, ln_ff_b + l * D_,
                                              ffa, nullptr);

        // FFN1: [512,96] x [2048,96] -> hbuf (relu)
        gemm(ffa, AD_, w116 + (size_t)l * FFN_ * DP, DP, b1 + (size_t)l * FFN_,
             hbuf, FFN_, T_ * B_, FFN_, DP, 1);

        // FFN2: [512,2048] x [80,2048] -> split-K over 8 chunks of 256, then reduce
        {
            dim3 grid(cdiv(D_, 32), (T_ * B_) / 16, SK_);
            wmma_gemm_splitk_kernel<<<grid, 32, 0, stream>>>(
                hbuf, FFN_, w216 + (size_t)l * DP * FFN_, FFN_,
                pbuf, AD_, T_ * B_, D_, FFN_ / SK_);
            splitk_reduce_kernel<<<cdiv(T_ * B_ * D_, 256), 256, 0, stream>>>(
                pbuf, b2 + l * D_, ffout, T_ * B_);
        }

        // x_next = LN(res + ffout)
        ln_kernel<<<T_ * B_, 32, 0, stream>>>(res, ffout,
                                              ln_o_g + l * D_, ln_o_b + l * D_,
                                              x, nullptr);
    }

    // -------- heads --------
    gemm(x, AD_, ctc16, DP, ctc_b, out_logits, V_, T_ * B_, V_, DP, 0);   // [T,B,V]
    encmean_kernel<<<cdiv(B_ * D_, 256), 256, 0, stream>>>(x, encm);
    gemm(encm, AD_, emb16, DP, emb_b, out_emb, EMB_, B_, EMB_, DP, 0);    // [B,EMB]
    memory_out_kernel<<<cdiv(B_ * M_ * D_, 256), 256, 0, stream>>>(memory, memcl, out_mem);
}
